// pointnet2_encoder2_76175539962310
// MI455X (gfx1250) — matrix operand order repaired and checked by reading, not compiled, not run on hardware
//
#include <hip/hip_runtime.h>
#include <hip/hip_bf16.h>
#include <stddef.h>

typedef __attribute__((ext_vector_type(16))) __bf16 v16bf;
typedef __attribute__((ext_vector_type(8)))  __bf16 v8bf;
typedef __attribute__((ext_vector_type(8)))  float  v8f;

#define EPSV 1e-5f

static const int BB = 8;
static const int NN = 4096;

// Packed bf16 activation layout (WMMA-B-native):
//   element (k, n) lives at Xb[ ((k>>5)*Ntot + n)*32 + (k&31) ]
// so a lane's 16-consecutive-K B-fragment is one contiguous 32-byte load.
__device__ __forceinline__ size_t pack_idx(int k, size_t n, size_t Ntot) {
    return ((((size_t)(k >> 5)) * Ntot + n) << 5) + (size_t)(k & 31);
}

// ---------------------------------------------------------------------------
// Farthest point sampling: one block per batch. dist kept in LDS.
// Matches jnp.argmax tie-breaking (lowest index wins among equal maxima).
// ---------------------------------------------------------------------------
#define FPS_T 256
__global__ void fps_kernel(const float* __restrict__ pts,   // [B][n][3]
                           int* __restrict__ out_idx,        // [B][npoint]
                           float* __restrict__ centers,      // [B][npoint][3]
                           int n, int npoint) {
    __shared__ float dist[4096];
    __shared__ float smax[FPS_T];
    __shared__ int   simax[FPS_T];
    __shared__ int   s_last;
    int b = blockIdx.x;
    int t = threadIdx.x;
    const float* P = pts + (size_t)b * n * 3;

    for (int j = t; j < n; j += FPS_T) dist[j] = 1e10f;
    if (t == 0) {
        s_last = 0;
        out_idx[(size_t)b * npoint] = 0;
        centers[((size_t)b * npoint) * 3 + 0] = P[0];
        centers[((size_t)b * npoint) * 3 + 1] = P[1];
        centers[((size_t)b * npoint) * 3 + 2] = P[2];
    }
    __syncthreads();

    for (int i = 1; i < npoint; ++i) {
        int last = s_last;
        float px = P[(size_t)last * 3 + 0];
        float py = P[(size_t)last * 3 + 1];
        float pz = P[(size_t)last * 3 + 2];
        float lmax = -1.0f; int lidx = 0;
        for (int j = t; j < n; j += FPS_T) {
            float dx = P[(size_t)j * 3 + 0] - px;
            float dy = P[(size_t)j * 3 + 1] - py;
            float dz = P[(size_t)j * 3 + 2] - pz;
            float d = dx * dx + dy * dy + dz * dz;
            float nd = fminf(dist[j], d);
            dist[j] = nd;
            if (nd > lmax) { lmax = nd; lidx = j; }
        }
        smax[t] = lmax; simax[t] = lidx;
        __syncthreads();
        for (int s = FPS_T / 2; s > 0; s >>= 1) {
            if (t < s) {
                bool take = (smax[t + s] > smax[t]) ||
                            (smax[t + s] == smax[t] && simax[t + s] < simax[t]);
                if (take) { smax[t] = smax[t + s]; simax[t] = simax[t + s]; }
            }
            __syncthreads();
        }
        if (t == 0) {
            int nxt = simax[0];
            s_last = nxt;
            out_idx[(size_t)b * npoint + i] = nxt;
            centers[((size_t)b * npoint + i) * 3 + 0] = P[(size_t)nxt * 3 + 0];
            centers[((size_t)b * npoint + i) * 3 + 1] = P[(size_t)nxt * 3 + 1];
            centers[((size_t)b * npoint + i) * 3 + 2] = P[(size_t)nxt * 3 + 2];
        }
        __syncthreads();
    }
}

// ---------------------------------------------------------------------------
// Ball query: ascending scan == reference's sort/take-first-k/pad-with-first.
// ---------------------------------------------------------------------------
__global__ void ball_query_kernel(const float* __restrict__ pts,     // [B][n][3]
                                  const float* __restrict__ centers, // [B][S][3]
                                  int* __restrict__ idx,             // [B][S][K]
                                  int n, int S, int K, float r2) {
    int g = blockIdx.x * blockDim.x + threadIdx.x;
    if (g >= BB * S) return;
    int s = g % S; int b = g / S;
    const float* P = pts + (size_t)b * n * 3;
    const float* c = centers + ((size_t)b * S + s) * 3;
    float cx = c[0], cy = c[1], cz = c[2];
    int* out = idx + ((size_t)b * S + s) * (size_t)K;
    int cnt = 0;
    for (int j = 0; j < n && cnt < K; ++j) {
        float dx = P[(size_t)j * 3 + 0] - cx;
        float dy = P[(size_t)j * 3 + 1] - cy;
        float dz = P[(size_t)j * 3 + 2] - cz;
        if (dx * dx + dy * dy + dz * dz < r2) out[cnt++] = j;
    }
    int first = (cnt > 0) ? out[0] : 0;
    for (; cnt < K; ++cnt) out[cnt] = first;
}

// ---------------------------------------------------------------------------
// Grouping: write packed bf16 activations. Channels:
//   0..2    -> relative xyz (point - center)
//   3..2+Cf -> gathered features feats[b][c][n]   (for SA1, feats = xyz^T)
//   rest    -> zero padding up to Kp (multiple of 32)
// For SA1 we pass feats==nullptr, Cf==3 and gather absolute xyz directly.
// ---------------------------------------------------------------------------
__global__ void group_pack_kernel(const float* __restrict__ pts,     // [B][n][3]
                                  const float* __restrict__ feats,   // [B][Cf][n] or null
                                  const float* __restrict__ centers, // [B][S][3]
                                  const int* __restrict__ idx,       // [B][S][K]
                                  __bf16* __restrict__ Xb,
                                  int n, int S, int K, int Cf, int Kp) {
    size_t Ntot = (size_t)BB * S * K;
    size_t g = (size_t)blockIdx.x * blockDim.x + threadIdx.x;
    if (g >= Ntot) return;
    size_t t = g / K;
    int s = (int)(t % S); int b = (int)(t / S);
    int j = idx[g];
    const float* p = pts + ((size_t)b * n + j) * 3;
    const float* c = centers + ((size_t)b * S + s) * 3;
    Xb[pack_idx(0, g, Ntot)] = (__bf16)(p[0] - c[0]);
    Xb[pack_idx(1, g, Ntot)] = (__bf16)(p[1] - c[1]);
    Xb[pack_idx(2, g, Ntot)] = (__bf16)(p[2] - c[2]);
    if (feats) {
        const float* F = feats + (size_t)b * Cf * n;
        for (int cc = 0; cc < Cf; ++cc)
            Xb[pack_idx(3 + cc, g, Ntot)] = (__bf16)F[(size_t)cc * n + j];
    } else {
        // SA1: features are the absolute xyz of the gathered point
        Xb[pack_idx(3, g, Ntot)] = (__bf16)p[0];
        Xb[pack_idx(4, g, Ntot)] = (__bf16)p[1];
        Xb[pack_idx(5, g, Ntot)] = (__bf16)p[2];
    }
    for (int cc = 3 + Cf; cc < Kp; ++cc)
        Xb[pack_idx(cc, g, Ntot)] = (__bf16)0.0f;
}

// Group-all for SA3: column n = b*S + s
__global__ void group_all_pack_kernel(const float* __restrict__ centers, // [B][S][3]
                                      const float* __restrict__ feats,   // [B][Cf][S]
                                      __bf16* __restrict__ Xb,
                                      int S, int Cf, int Kp) {
    size_t Ntot = (size_t)BB * S;
    int g = blockIdx.x * blockDim.x + threadIdx.x;
    if (g >= (int)Ntot) return;
    int s = g % S; int b = g / S;
    const float* c = centers + ((size_t)b * S + s) * 3;
    Xb[pack_idx(0, g, Ntot)] = (__bf16)c[0];
    Xb[pack_idx(1, g, Ntot)] = (__bf16)c[1];
    Xb[pack_idx(2, g, Ntot)] = (__bf16)c[2];
    for (int cc = 0; cc < Cf; ++cc)
        Xb[pack_idx(3 + cc, g, Ntot)] = (__bf16)feats[((size_t)b * Cf + cc) * S + s];
    for (int cc = 3 + Cf; cc < Kp; ++cc)
        Xb[pack_idx(cc, g, Ntot)] = (__bf16)0.0f;
}

// ---------------------------------------------------------------------------
// Weight conversion f32 -> bf16, zero-padded to [Mpad x Kpad] row-major.
// ---------------------------------------------------------------------------
__global__ void convert_weights_kernel(const float* __restrict__ W, // [M x Cin]
                                       __bf16* __restrict__ Wb,     // [Mpad x Kpad]
                                       int M, int Cin, int Mpad, int Kpad) {
    int g = blockIdx.x * blockDim.x + threadIdx.x;
    if (g >= Mpad * Kpad) return;
    int k = g % Kpad; int m = g / Kpad;
    float v = (m < M && k < Cin) ? W[(size_t)m * Cin + k] : 0.0f;
    Wb[g] = (__bf16)v;
}

// ---------------------------------------------------------------------------
// WMMA GEMM: Y[m][n] = sum_k Wb[m][k] * X[k][n] + bias[m]
// 256-thread blocks (8 waves). Each wave computes a 16x64 strip:
// one A fragment reused across 4 B fragments / 4 accumulators.
// Inner loop per K-step: 2 x b128 A loads, 4 x 32B B loads, 4 x v_wmma.
// No branches in the loop -> EXEC stays all-ones (WMMA requirement).
// Fragment layouts per CDNA5 ISA 7.12.2 (wave32):
//   A (16-bit 16x32): row = l15; lanes 0-15 hold K {0..7,16..23},
//                     lanes 16-31 hold K {8..15,24..31}
//   B (16-bit 32x16): col = l15; code uses lanes 0-15 K 0..15, lanes 16-31 K 16..31, as for A
//   C/D: m = vgpr + 8*(lane>=16), col = l15
// ---------------------------------------------------------------------------
__global__ __launch_bounds__(256)
void wmma_gemm_bias_kernel(const __bf16* __restrict__ Wb,  // [Mpad x Kpad]
                           const __bf16* __restrict__ Xb,  // packed [Kpad x Ntot]
                           const float* __restrict__ bias, // [M]
                           float* __restrict__ Y,          // [M x Ntot]
                           int M, int Kpad, int Ntot) {
    const int lane = threadIdx.x & 31;
    const int wave = threadIdx.x >> 5;
    const int l15  = lane & 15;
    const int half = lane >> 4;
    const int nbase = (blockIdx.x * 8 + wave) * 64;
    const int mbase = blockIdx.y * 16;

    v8f acc[4];
    acc[0] = (v8f){}; acc[1] = (v8f){}; acc[2] = (v8f){}; acc[3] = (v8f){};

    const __bf16* wrow = Wb + (size_t)(mbase + l15) * Kpad;

    for (int kb = 0; kb < Kpad; kb += 32) {
        // A fragment: 16 contiguous K (kb + 16*half .. +15), matching B
        v8bf alo = *(const v8bf*)(wrow + kb + (half << 4));
        v8bf ahi = *(const v8bf*)(wrow + kb + (half << 4) + 8);
        v16bf a;
        #pragma unroll
        for (int e = 0; e < 8; ++e) { a[e] = alo[e]; a[e + 8] = ahi[e]; }

        const __bf16* xt = Xb + ((((size_t)(kb >> 5)) * Ntot) << 5) + (half << 4);
        #pragma unroll
        for (int t = 0; t < 4; ++t) {
            int col = nbase + t * 16 + l15;
            v16bf b = *(const v16bf*)(xt + ((size_t)col << 5));
            acc[t] = __builtin_amdgcn_wmma_f32_16x16x32_bf16(
                false, a, false, b, (short)0, acc[t], false, false);
        }
    }

    #pragma unroll
    for (int v = 0; v < 8; ++v) {
        int m = mbase + v + half * 8;
        if (m < M) {
            float bi = bias[m];
            #pragma unroll
            for (int t = 0; t < 4; ++t)
                Y[(size_t)m * Ntot + (nbase + t * 16 + l15)] = acc[t][v] + bi;
        }
    }
}

// ---------------------------------------------------------------------------
// BatchNorm statistics (population mean/var over all N columns per channel)
// ---------------------------------------------------------------------------
__global__ void zero_kernel(float* p, int n) {
    int g = blockIdx.x * blockDim.x + threadIdx.x;
    if (g < n) p[g] = 0.0f;
}

__global__ void bn_stats_kernel(const float* __restrict__ Y,
                                float* __restrict__ sum,
                                float* __restrict__ sumsq,
                                size_t Ntot) {
    __shared__ float ssum[256], ssq[256];
    int m = blockIdx.y;
    float s = 0.0f, q = 0.0f;
    for (size_t n = (size_t)blockIdx.x * 256 + threadIdx.x; n < Ntot;
         n += (size_t)gridDim.x * 256) {
        float v = Y[(size_t)m * Ntot + n];
        s += v; q += v * v;
    }
    ssum[threadIdx.x] = s; ssq[threadIdx.x] = q;
    __syncthreads();
    for (int st = 128; st > 0; st >>= 1) {
        if (threadIdx.x < st) {
            ssum[threadIdx.x] += ssum[threadIdx.x + st];
            ssq[threadIdx.x]  += ssq[threadIdx.x + st];
        }
        __syncthreads();
    }
    if (threadIdx.x == 0) {
        atomicAdd(&sum[m], ssum[0]);
        atomicAdd(&sumsq[m], ssq[0]);
    }
}

// Normalize + ReLU in-place (f32) and write packed bf16 for the next layer.
// Grid covers Kp*Ntot; rows >= M write zero padding into the packed buffer.
__global__ void bn_apply_pack_kernel(float* __restrict__ Y,
                                     __bf16* __restrict__ Xb,
                                     const float* __restrict__ sum,
                                     const float* __restrict__ sumsq,
                                     const float* __restrict__ gamma,
                                     const float* __restrict__ beta,
                                     int M, size_t Ntot, size_t totalKp) {
    size_t g = (size_t)blockIdx.x * blockDim.x + threadIdx.x;
    if (g >= totalKp) return;
    int m = (int)(g / Ntot);
    size_t n = g % Ntot;
    size_t pidx = pack_idx(m, n, Ntot);
    if (m < M) {
        float inv = 1.0f / (float)Ntot;
        float mean = sum[m] * inv;
        float var = sumsq[m] * inv - mean * mean;
        float v = gamma[m] * (Y[g] - mean) * rsqrtf(var + EPSV) + beta[m];
        v = v > 0.0f ? v : 0.0f;
        Y[g] = v;
        Xb[pidx] = (__bf16)v;
    } else {
        Xb[pidx] = (__bf16)0.0f;
    }
}

// ---------------------------------------------------------------------------
// Max pool over K: pooled[b][cOff+m][s] = max_k Y[m][((b*S+s)*K+k)]
// ---------------------------------------------------------------------------
__global__ void maxpool_kernel(const float* __restrict__ Y,
                               float* __restrict__ pooled,
                               int M, int S, int K, int cOff, int Ctot) {
    int g = blockIdx.x * blockDim.x + threadIdx.x;
    if (g >= BB * M * S) return;
    int s = g % S; int t = g / S; int m = t % M; int b = t / M;
    size_t Ntot = (size_t)BB * S * K;
    const float* row = Y + (size_t)m * Ntot + ((size_t)b * S + s) * K;
    float mx = row[0];
    for (int k = 1; k < K; ++k) mx = fmaxf(mx, row[k]);
    pooled[((size_t)b * Ctot + cOff + m) * S + s] = mx;
}

// Final: out[b][m] = max_s Y[m][b*S+s]
__global__ void final_max_kernel(const float* __restrict__ Y,
                                 float* __restrict__ out, int M, int S) {
    int g = blockIdx.x * blockDim.x + threadIdx.x;
    if (g >= BB * M) return;
    int m = g % M; int b = g / M;
    size_t Ntot = (size_t)BB * S;
    const float* row = Y + (size_t)m * Ntot + (size_t)b * S;
    float mx = row[0];
    for (int s = 1; s < S; ++s) mx = fmaxf(mx, row[s]);
    out[(size_t)b * M + m] = mx;
}

// ---------------------------------------------------------------------------
// Host-side MLP runner. Input activations already packed bf16 in Xb.
// After each layer: f32 result in Y (for stats / pooling), next-layer packed
// bf16 input regenerated into Xb by bn_apply_pack.
// ---------------------------------------------------------------------------
struct LayerP { const float *w, *b, *g, *be; };

static void run_mlp(const LayerP* L, const int* dims, int nl,
                    __bf16* Xb, float* Y, __bf16* wbuf, float* stats,
                    size_t Ntot, hipStream_t stream) {
    for (int i = 0; i < nl; ++i) {
        int Cin = dims[i], M = dims[i + 1];
        int Kpad = (Cin + 31) & ~31;
        int Mpad = (M + 15) & ~15;
        int wtot = Mpad * Kpad;
        convert_weights_kernel<<<(wtot + 255) / 256, 256, 0, stream>>>(
            L[i].w, wbuf, M, Cin, Mpad, Kpad);
        dim3 grid((unsigned)(Ntot / 512), (unsigned)(Mpad / 16));
        wmma_gemm_bias_kernel<<<grid, 256, 0, stream>>>(
            wbuf, Xb, L[i].b, Y, M, Kpad, (int)Ntot);
        zero_kernel<<<(2048 + 255) / 256, 256, 0, stream>>>(stats, 2048);
        dim3 sg(64, (unsigned)M);
        bn_stats_kernel<<<sg, 256, 0, stream>>>(Y, stats, stats + 1024, Ntot);
        int Kp_out = (M + 31) & ~31;
        size_t totalKp = (size_t)Kp_out * Ntot;
        bn_apply_pack_kernel<<<(unsigned)((totalKp + 255) / 256), 256, 0, stream>>>(
            Y, Xb, stats, stats + 1024, L[i].g, L[i].be, M, Ntot, totalKp);
    }
}

// ---------------------------------------------------------------------------
extern "C" void kernel_launch(void* const* d_in, const int* in_sizes, int n_in,
                              void* d_out, int out_size, void* d_ws, size_t ws_size,
                              hipStream_t stream) {
    (void)in_sizes; (void)n_in; (void)out_size; (void)ws_size;
    const float* xyz = (const float*)d_in[0];

    // --- workspace carve-out ---
    char* ws = (char*)d_ws;
    size_t off = 0;
    auto carve = [&](size_t bytes) -> void* {
        void* p = ws + off;
        off += (bytes + 255) & ~(size_t)255;
        return p;
    };
    int*   fps1    = (int*)  carve((size_t)BB * 512 * 4);
    float* c1      = (float*)carve((size_t)BB * 512 * 3 * 4);
    int*   fps2    = (int*)  carve((size_t)BB * 128 * 4);
    float* c2      = (float*)carve((size_t)BB * 128 * 3 * 4);
    int*   idxb    = (int*)  carve((size_t)BB * 512 * 128 * 4);
    float* pooled1 = (float*)carve((size_t)BB * 384 * 512 * 4);
    float* pooled2 = (float*)carve((size_t)BB * 512 * 128 * 4);
    float* stats   = (float*)carve((size_t)2048 * 4);
    __bf16* wbuf   = (__bf16*)carve((size_t)1024 * 544 * 2);
    // packed bf16 activations: max Kpad*Ntot = 416*131072 vs 128*524288 -> 67.1M
    __bf16* Xb     = (__bf16*)carve((size_t)67108864 * 2);
    // f32 layer output: max M*Ntot = 128*524288 = 67.1M floats
    float* Y       = (float*)carve((size_t)67108864 * 4);

    // --- parameter walker (setup_inputs order: sa1[3][3], sa2[2][3], sa3[3];
    //     each layer contributes w, b, gamma, beta) ---
    int pi = 1;
    auto nextLayer = [&](LayerP& Lp) {
        Lp.w  = (const float*)d_in[pi++];
        Lp.b  = (const float*)d_in[pi++];
        Lp.g  = (const float*)d_in[pi++];
        Lp.be = (const float*)d_in[pi++];
    };

    // ================= SA1 (MSG) =================
    fps_kernel<<<BB, FPS_T, 0, stream>>>(xyz, fps1, c1, NN, 512);

    const int   K1[3]    = {32, 64, 128};
    const float R1[3]    = {0.1f, 0.2f, 0.4f};
    const int   dims1[4] = {6, 64, 96, 128};
    for (int br = 0; br < 3; ++br) {
        LayerP L[3];
        nextLayer(L[0]); nextLayer(L[1]); nextLayer(L[2]);
        size_t Ntot = (size_t)BB * 512 * K1[br];
        ball_query_kernel<<<(BB * 512 + 255) / 256, 256, 0, stream>>>(
            xyz, c1, idxb, NN, 512, K1[br], R1[br] * R1[br]);
        group_pack_kernel<<<(unsigned)((Ntot + 255) / 256), 256, 0, stream>>>(
            xyz, nullptr, c1, idxb, Xb, NN, 512, K1[br], 3, 32);
        run_mlp(L, dims1, 3, Xb, Y, wbuf, stats, Ntot, stream);
        maxpool_kernel<<<(BB * 128 * 512 + 255) / 256, 256, 0, stream>>>(
            Y, pooled1, 128, 512, K1[br], br * 128, 384);
    }

    // ================= SA2 (MSG) =================
    fps_kernel<<<BB, FPS_T, 0, stream>>>(c1, fps2, c2, 512, 128);

    const int   K2[2]    = {64, 128};
    const float R2[2]    = {0.4f, 0.8f};
    const int   dims2[4] = {387, 128, 196, 256};
    for (int br = 0; br < 2; ++br) {
        LayerP L[3];
        nextLayer(L[0]); nextLayer(L[1]); nextLayer(L[2]);
        size_t Ntot = (size_t)BB * 128 * K2[br];
        ball_query_kernel<<<(BB * 128 + 255) / 256, 256, 0, stream>>>(
            c1, c2, idxb, 512, 128, K2[br], R2[br] * R2[br]);
        group_pack_kernel<<<(unsigned)((Ntot + 255) / 256), 256, 0, stream>>>(
            c1, pooled1, c2, idxb, Xb, 512, 128, K2[br], 384, 416);
        run_mlp(L, dims2, 3, Xb, Y, wbuf, stats, Ntot, stream);
        maxpool_kernel<<<(BB * 256 * 128 + 255) / 256, 256, 0, stream>>>(
            Y, pooled2, 256, 128, K2[br], br * 256, 512);
    }

    // ================= SA3 (group-all) =================
    {
        LayerP L[3];
        nextLayer(L[0]); nextLayer(L[1]); nextLayer(L[2]);
        const int dims3[4] = {515, 512, 512, 1024};
        group_all_pack_kernel<<<(BB * 128 + 255) / 256, 256, 0, stream>>>(
            c2, pooled2, Xb, 128, 512, 544);
        size_t Ntot = (size_t)BB * 128;  // 1024 columns
        run_mlp(L, dims3, 3, Xb, Y, wbuf, stats, Ntot, stream);
        final_max_kernel<<<(BB * 1024 + 255) / 256, 256, 0, stream>>>(
            Y, (float*)d_out, 1024, 128);
    }
}
